// StereoNet_55774445306161
// MI455X (gfx1250) — matrix operand order repaired and checked by reading, not compiled, not run on hardware
//
#include <hip/hip_runtime.h>
#include <hip/hip_bf16.h>

// ---------------------------------------------------------------------------
// GATv2 x2 + MLP head for MI455X (gfx1250).
// Dense transforms run as bf16 WMMA GEMMs (v_wmma_f32_16x16x32_bf16) with
// NB=4 column-tile register blocking: all tile loads for a k-step are issued
// before the 4 WMMAs so waits can be partial (load/compute overlap). Edge
// gather/softmax/scatter are VALU kernels (gather tables L2-resident:
// 41MB << 192MB L2).
// ---------------------------------------------------------------------------

#define NNODES 20000
#define NEDGES 320000
#define NODE_DIM 64
#define EDGE_DIM 32
#define HIDC 128
#define NHEADS 4
#define HC1 (NHEADS * HIDC) // 512

typedef __attribute__((ext_vector_type(16))) __bf16 v16bf;
typedef __attribute__((ext_vector_type(8)))  float  v8f;

// ---------------- WMMA tile loaders (wave32, 16-bit element layouts) -------
// A tile: 16x32 (MxK), row-major with leading dim ld.
// Layout (ISA 7.12.2): lanes 0-15 = rows, elements 0..7 -> K=half*8..+7,
// elements 8..15 -> K=16+half*8..+7 where half = lane>>4.
__device__ __forceinline__ v16bf load_tile_A(const __bf16* tile, int ld) {
  int lane = threadIdx.x & 31;
  int row  = lane & 15;
  int half = lane >> 4;
  const __bf16* p = tile + (size_t)row * ld + half * 8;
  union { v16bf v; uint4 q[2]; } u;
  u.q[0] = *(const uint4*)(p);        // K = half*8 .. half*8+7
  u.q[1] = *(const uint4*)(p + 16);   // K = 16+half*8 ..
  return u.v;
}
// B tile: 32x16 (KxN) consumed from Bt stored N-major (Bt[n][k], ld = K).
// Layout: same K order as A: elements 0..7 -> K=half*8..+7, 8..15 -> K=16+half*8..+7.
__device__ __forceinline__ v16bf load_tile_B(const __bf16* tile, int ld) {
  int lane = threadIdx.x & 31;
  int n    = lane & 15;
  int half = lane >> 4;
  const __bf16* p = tile + (size_t)n * ld + half * 8;
  union { v16bf v; uint4 q[2]; } u;
  u.q[0] = *(const uint4*)(p);
  u.q[1] = *(const uint4*)(p + 16);
  return u.v;
}

// ---------------- WMMA GEMM: C[M,N] = A[M,K] * Bt[N,K]^T (+bias, relu) -----
// 4 waves per block, each wave owns one 16-row M tile and NB 16-col N tiles.
// grid.x = Ntiles/NB, grid.y = ceil(Mtiles/4). C and/or Cbf may be null.
template <int NB>
__global__ void __launch_bounds__(128)
wmma_gemm_bf16(const __bf16* __restrict__ A, const __bf16* __restrict__ Bt,
               const float* __restrict__ bias, float* __restrict__ C,
               __bf16* __restrict__ Cbf, int K, int ldc, int nvalid, int relu,
               int Mtiles) {
  int mtile = blockIdx.y * 4 + (threadIdx.x >> 5);
  if (mtile >= Mtiles) return;            // wave-uniform: EXEC stays all-ones
  int ntile0 = blockIdx.x * NB;
  const __bf16* Ab = A + (size_t)mtile * 16 * K;
  v8f acc[NB] = {};
  for (int k0 = 0; k0 < K; k0 += 32) {
    // issue ALL loads for this k-step first, then the WMMAs -> partial waits
    v16bf a = load_tile_A(Ab + k0, K);
    v16bf b[NB];
#pragma unroll
    for (int j = 0; j < NB; ++j)
      b[j] = load_tile_B(Bt + (size_t)(ntile0 + j) * 16 * K + k0, K);
#pragma unroll
    for (int j = 0; j < NB; ++j)
      acc[j] = __builtin_amdgcn_wmma_f32_16x16x32_bf16(
          false, a, false, b[j], (short)0, acc[j], false, false);
  }
  int lane = threadIdx.x & 31;
  int col  = lane & 15;
  int rbase = (lane >> 4) * 8;   // C layout: VGPR e -> row e (lanes<16) / e+8
#pragma unroll
  for (int j = 0; j < NB; ++j) {
    int gcol = (ntile0 + j) * 16 + col;
    if (gcol >= nvalid) continue;
    float bv = bias ? bias[gcol] : 0.f;
#pragma unroll
    for (int e = 0; e < 8; ++e) {
      int grow = mtile * 16 + rbase + e;
      float v = acc[j][e] + bv;
      if (relu) v = v > 0.f ? v : 0.f;
      size_t o = (size_t)grow * ldc + gcol;
      if (C)   C[o]   = v;
      if (Cbf) Cbf[o] = (__bf16)v;
    }
  }
}

// ---------------- helpers ---------------------------------------------------
__global__ void fill_f32(float* p, float v, long long n) {
  long long t = blockIdx.x * (long long)blockDim.x + threadIdx.x;
  if (t < n) p[t] = v;
}
__global__ void f32_to_bf16(const float* __restrict__ in, __bf16* __restrict__ out,
                            long long n) {
  long long t = blockIdx.x * (long long)blockDim.x + threadIdx.x;
  if (t < n) out[t] = (__bf16)in[t];
}
// Wt[n*K+k] = (n < Ncols) ? W[k*Ncols+n] : 0   (W is [K,Ncols] row-major)
__global__ void transpose_pack(const float* __restrict__ W, __bf16* __restrict__ Wt,
                               int K, int Ncols, int Npad) {
  long long t = blockIdx.x * (long long)blockDim.x + threadIdx.x;
  long long total = (long long)Npad * K;
  if (t >= total) return;
  int n = (int)(t / K), k = (int)(t % K);
  Wt[t] = (n < Ncols) ? (__bf16)W[(size_t)k * Ncols + n] : (__bf16)0.f;
}

__device__ __forceinline__ void atomicMaxF(float* addr, float val) {
  // works for all floats given -inf init (0xFF800000)
  if (val >= 0.f) atomicMax((int*)addr, __float_as_int(val));
  else            atomicMin((unsigned int*)addr, __float_as_uint(val));
}

// ---------------- edge kernels ---------------------------------------------
// logits[e,h] = att[h,:] . leaky_relu(xl[src,h,:] + xr[dst,h,:] + ee[e,h,:])
__global__ void edge_logits_k(const float* __restrict__ xl, const float* __restrict__ xr,
                              const __bf16* __restrict__ ee,
                              const int* __restrict__ src, const int* __restrict__ dst,
                              const float* __restrict__ att,
                              float* __restrict__ logits, float* __restrict__ nmax,
                              int E, int H, int C) {
  int t = blockIdx.x * blockDim.x + threadIdx.x;
  if (t >= E * H) return;
  int e = t / H, h = t - e * H;
  int s = src[e], d = dst[e];
  const float*  pl = xl + ((size_t)s * H + h) * C;
  const float*  pr = xr + ((size_t)d * H + h) * C;
  const __bf16* pe = ee + ((size_t)e * H + h) * C;
  const float*  pa = att + (size_t)h * C;
  float acc = 0.f;
  for (int c = 0; c < C; c += 4) {
    float4 a = *(const float4*)(pl + c);
    float4 b = *(const float4*)(pr + c);
    float4 w = *(const float4*)(pa + c);
    float m;
    m = a.x + b.x + (float)pe[c + 0]; m = m > 0.f ? m : 0.2f * m; acc += w.x * m;
    m = a.y + b.y + (float)pe[c + 1]; m = m > 0.f ? m : 0.2f * m; acc += w.y * m;
    m = a.z + b.z + (float)pe[c + 2]; m = m > 0.f ? m : 0.2f * m; acc += w.z * m;
    m = a.w + b.w + (float)pe[c + 3]; m = m > 0.f ? m : 0.2f * m; acc += w.w * m;
  }
  logits[t] = acc;
  atomicMaxF(&nmax[(size_t)d * H + h], acc);
}

__global__ void edge_expdenom_k(const float* __restrict__ logits,
                                const float* __restrict__ nmax,
                                const int* __restrict__ dst,
                                float* __restrict__ ex, float* __restrict__ den,
                                int E, int H) {
  int t = blockIdx.x * blockDim.x + threadIdx.x;
  if (t >= E * H) return;
  int e = t / H, h = t - e * H;
  int d = dst[e];
  float v = __expf(logits[t] - nmax[(size_t)d * H + h]);
  ex[t] = v;
  atomicAdd(&den[(size_t)d * H + h], v);
}

// out[dst,h,:] += (ex[e,h]/den[dst,h]) * xl[src,h,:], 4 channels per thread
__global__ void edge_scatter_k(const float* __restrict__ xl,
                               const int* __restrict__ src, const int* __restrict__ dst,
                               const float* __restrict__ ex, const float* __restrict__ den,
                               float* __restrict__ out, int E, int H, int C) {
  int chunks = (H * C) >> 2;
  long long t = blockIdx.x * (long long)blockDim.x + threadIdx.x;
  if (t >= (long long)E * chunks) return;
  int e  = (int)(t / chunks);
  int j  = (int)(t - (long long)e * chunks);
  int cb = j << 2;
  int h  = cb / C;
  int s = src[e], d = dst[e];
  float alpha = ex[(size_t)e * H + h] / den[(size_t)d * H + h];
  float4 v = *(const float4*)(xl + (size_t)s * H * C + cb);
  float* po = out + (size_t)d * H * C + cb;
  atomicAdd(po + 0, alpha * v.x);
  atomicAdd(po + 1, alpha * v.y);
  atomicAdd(po + 2, alpha * v.z);
  atomicAdd(po + 3, alpha * v.w);
}

__global__ void bias_act_k(float* __restrict__ h, const float* __restrict__ bias,
                           __bf16* __restrict__ hb, int HC, long long total, int relu) {
  long long t = blockIdx.x * (long long)blockDim.x + threadIdx.x;
  if (t >= total) return;
  int j = (int)(t % HC);
  float v = h[t] + bias[j];
  if (relu) v = v > 0.f ? v : 0.f;
  h[t] = v;
  if (hb) hb[t] = (__bf16)v;
}

// ---------------- launcher --------------------------------------------------
extern "C" void kernel_launch(void* const* d_in, const int* in_sizes, int n_in,
                              void* d_out, int out_size, void* d_ws, size_t ws_size,
                              hipStream_t stream) {
  const int N = NNODES, E = NEDGES;
  const int MT_N = N / 16;         // 1250 node M-tiles
  const int MT_E = E / 16;         // 20000 edge M-tiles
  const float* x    = (const float*)d_in[0];
  const int*   ei   = (const int*)d_in[1];      // [2,E] int32 (jax default x64-off)
  const float* ea   = (const float*)d_in[2];
  const float* Wl1  = (const float*)d_in[3];  const float* bl1  = (const float*)d_in[4];
  const float* Wr1  = (const float*)d_in[5];  const float* br1  = (const float*)d_in[6];
  const float* We1  = (const float*)d_in[7];
  const float* att1 = (const float*)d_in[8];  const float* bias1 = (const float*)d_in[9];
  const float* Wl2  = (const float*)d_in[10]; const float* bl2  = (const float*)d_in[11];
  const float* Wr2  = (const float*)d_in[12]; const float* br2  = (const float*)d_in[13];
  const float* We2  = (const float*)d_in[14];
  const float* att2 = (const float*)d_in[15]; const float* bias2 = (const float*)d_in[16];
  const float* Wh1  = (const float*)d_in[17]; const float* bh1  = (const float*)d_in[18];
  const float* Wh2  = (const float*)d_in[19]; const float* bh2  = (const float*)d_in[20];
  const int* src = ei;
  const int* dst = ei + E;

  // --- bump allocator over d_ws (256B aligned) ---
  char* ws = (char*)d_ws; size_t off = 0;
  auto alloc = [&](size_t bytes) -> void* {
    void* p = ws + off; off += (bytes + 255) & ~(size_t)255; return p;
  };
  __bf16* xh   = (__bf16*)alloc((size_t)N * NODE_DIM * 2);
  __bf16* eah  = (__bf16*)alloc((size_t)E * EDGE_DIM * 2);
  __bf16* Wl1t = (__bf16*)alloc((size_t)HC1 * NODE_DIM * 2);
  __bf16* Wr1t = (__bf16*)alloc((size_t)HC1 * NODE_DIM * 2);
  __bf16* We1t = (__bf16*)alloc((size_t)HC1 * EDGE_DIM * 2);
  __bf16* Wl2t = (__bf16*)alloc((size_t)HIDC * HC1 * 2);
  __bf16* Wr2t = (__bf16*)alloc((size_t)HIDC * HC1 * 2);
  __bf16* We2t = (__bf16*)alloc((size_t)HIDC * EDGE_DIM * 2);
  __bf16* Wh1t = (__bf16*)alloc((size_t)64 * HIDC * 2);
  __bf16* Wh2t = (__bf16*)alloc((size_t)16 * 64 * 2);        // padded 2 -> 16 cols
  float*  xl1  = (float*)alloc((size_t)N * HC1 * 4);
  float*  xr1  = (float*)alloc((size_t)N * HC1 * 4);
  __bf16* eeB  = (__bf16*)alloc((size_t)E * HC1 * 2);        // reused layer2 (E*128)
  float*  lg   = (float*)alloc((size_t)E * NHEADS * 4);      // reused layer2 (E)
  float*  exb  = (float*)alloc((size_t)E * NHEADS * 4);
  float*  nmax = (float*)alloc((size_t)N * NHEADS * 4);
  float*  den  = (float*)alloc((size_t)N * NHEADS * 4);
  float*  g1   = (float*)alloc((size_t)N * HC1 * 4);         // layer1 out (in-place relu)
  __bf16* h1h  = (__bf16*)alloc((size_t)N * HC1 * 2);
  float*  xl2  = (float*)alloc((size_t)N * HIDC * 4);
  float*  xr2  = (float*)alloc((size_t)N * HIDC * 4);
  float*  g2   = (float*)alloc((size_t)N * HIDC * 4);
  __bf16* h2h  = (__bf16*)alloc((size_t)N * HIDC * 2);
  float*  h3   = (float*)alloc((size_t)N * 64 * 4);
  __bf16* h3h  = (__bf16*)alloc((size_t)N * 64 * 2);
  (void)ws_size; (void)in_sizes; (void)n_in; (void)out_size;

  auto nb = [](long long n) { return (unsigned)((n + 255) / 256); };
  auto mg = [](int mt) { return (unsigned)((mt + 3) / 4); };  // M-tile blocks (4 waves)

  // --- pack inputs / weights to bf16 ---
  f32_to_bf16<<<nb((long long)N * NODE_DIM), 256, 0, stream>>>(x,  xh,  (long long)N * NODE_DIM);
  f32_to_bf16<<<nb((long long)E * EDGE_DIM), 256, 0, stream>>>(ea, eah, (long long)E * EDGE_DIM);
  transpose_pack<<<nb((long long)HC1 * NODE_DIM), 256, 0, stream>>>(Wl1, Wl1t, NODE_DIM, HC1, HC1);
  transpose_pack<<<nb((long long)HC1 * NODE_DIM), 256, 0, stream>>>(Wr1, Wr1t, NODE_DIM, HC1, HC1);
  transpose_pack<<<nb((long long)HC1 * EDGE_DIM), 256, 0, stream>>>(We1, We1t, EDGE_DIM, HC1, HC1);
  transpose_pack<<<nb((long long)HIDC * HC1), 256, 0, stream>>>(Wl2, Wl2t, HC1, HIDC, HIDC);
  transpose_pack<<<nb((long long)HIDC * HC1), 256, 0, stream>>>(Wr2, Wr2t, HC1, HIDC, HIDC);
  transpose_pack<<<nb((long long)HIDC * EDGE_DIM), 256, 0, stream>>>(We2, We2t, EDGE_DIM, HIDC, HIDC);
  transpose_pack<<<nb((long long)64 * HIDC), 256, 0, stream>>>(Wh1, Wh1t, HIDC, 64, 64);
  transpose_pack<<<nb((long long)16 * 64), 256, 0, stream>>>(Wh2, Wh2t, 64, 2, 16);

  // ================= Layer 1 (heads=4, C=128) =================
  dim3 blk(128);
  wmma_gemm_bf16<4><<<dim3(HC1 / 64, mg(MT_N)), blk, 0, stream>>>(xh, Wl1t, bl1, xl1, nullptr, NODE_DIM, HC1, HC1, 0, MT_N);
  wmma_gemm_bf16<4><<<dim3(HC1 / 64, mg(MT_N)), blk, 0, stream>>>(xh, Wr1t, br1, xr1, nullptr, NODE_DIM, HC1, HC1, 0, MT_N);
  wmma_gemm_bf16<4><<<dim3(HC1 / 64, mg(MT_E)), blk, 0, stream>>>(eah, We1t, nullptr, nullptr, eeB, EDGE_DIM, HC1, HC1, 0, MT_E);

  fill_f32<<<nb((long long)N * NHEADS), 256, 0, stream>>>(nmax, -INFINITY, (long long)N * NHEADS);
  fill_f32<<<nb((long long)N * NHEADS), 256, 0, stream>>>(den, 0.f, (long long)N * NHEADS);
  fill_f32<<<nb((long long)N * HC1), 256, 0, stream>>>(g1, 0.f, (long long)N * HC1);

  edge_logits_k<<<nb((long long)E * NHEADS), 256, 0, stream>>>(xl1, xr1, eeB, src, dst, att1, lg, nmax, E, NHEADS, HIDC);
  edge_expdenom_k<<<nb((long long)E * NHEADS), 256, 0, stream>>>(lg, nmax, dst, exb, den, E, NHEADS);
  edge_scatter_k<<<nb((long long)E * (HC1 / 4)), 256, 0, stream>>>(xl1, src, dst, exb, den, g1, E, NHEADS, HIDC);
  bias_act_k<<<nb((long long)N * HC1), 256, 0, stream>>>(g1, bias1, h1h, HC1, (long long)N * HC1, 1);

  // ================= Layer 2 (heads=1, C=128) =================
  wmma_gemm_bf16<4><<<dim3(HIDC / 64, mg(MT_N)), blk, 0, stream>>>(h1h, Wl2t, bl2, xl2, nullptr, HC1, HIDC, HIDC, 0, MT_N);
  wmma_gemm_bf16<4><<<dim3(HIDC / 64, mg(MT_N)), blk, 0, stream>>>(h1h, Wr2t, br2, xr2, nullptr, HC1, HIDC, HIDC, 0, MT_N);
  wmma_gemm_bf16<4><<<dim3(HIDC / 64, mg(MT_E)), blk, 0, stream>>>(eah, We2t, nullptr, nullptr, eeB, EDGE_DIM, HIDC, HIDC, 0, MT_E);

  fill_f32<<<nb((long long)N), 256, 0, stream>>>(nmax, -INFINITY, (long long)N);
  fill_f32<<<nb((long long)N), 256, 0, stream>>>(den, 0.f, (long long)N);
  fill_f32<<<nb((long long)N * HIDC), 256, 0, stream>>>(g2, 0.f, (long long)N * HIDC);

  edge_logits_k<<<nb((long long)E), 256, 0, stream>>>(xl2, xr2, eeB, src, dst, att2, lg, nmax, E, 1, HIDC);
  edge_expdenom_k<<<nb((long long)E), 256, 0, stream>>>(lg, nmax, dst, exb, den, E, 1);
  edge_scatter_k<<<nb((long long)E * (HIDC / 4)), 256, 0, stream>>>(xl2, src, dst, exb, den, g2, E, 1, HIDC);
  bias_act_k<<<nb((long long)N * HIDC), 256, 0, stream>>>(g2, bias2, h2h, HIDC, (long long)N * HIDC, 0);

  // ================= MLP head =================
  wmma_gemm_bf16<4><<<dim3(1, mg(MT_N)), blk, 0, stream>>>(h2h, Wh1t, bh1, h3, h3h, HIDC, 64, 64, 1, MT_N);
  wmma_gemm_bf16<1><<<dim3(1, mg(MT_N)), blk, 0, stream>>>(h3h, Wh2t, bh2, (float*)d_out, nullptr, 64, 2, 2, 0, MT_N);
}
